// Model_28853590294623
// MI455X (gfx1250) — matrix operand order repaired and checked by reading, not compiled, not run on hardware
//
#include <hip/hip_runtime.h>
#include <hip/hip_bf16.h>

typedef __attribute__((ext_vector_type(16))) _Float16 v16h;
typedef __attribute__((ext_vector_type(2)))  _Float16 h2;
typedef __attribute__((ext_vector_type(8)))  float    v8f;
typedef _Float16 half_t;

// Problem constants (fixed by the reference's setup_inputs)
#define N0 262144
#define N1 65536
#define N2 16384
#define HDIM 128
#define CDIM 47
#define CPAD 48

// ---------------------------------------------------------------- utilities

__global__ void zero_f32(float* __restrict__ p, int n) {
  int i = blockIdx.x * blockDim.x + threadIdx.x;
  if (i < n) p[i] = 0.0f;
}

// W1hT[n*128+k] = (f16) W1[k*128+n]   (transposed so K is contiguous per column)
__global__ void conv_w1(const float* __restrict__ W1, half_t* __restrict__ W1hT) {
  int i = blockIdx.x * blockDim.x + threadIdx.x;   // 128*128
  int k = i & 127, n = i >> 7;
  W1hT[n * HDIM + k] = (half_t)W1[k * HDIM + n];
}

// W2hT[n*128+k] = (f16) W2[k*47+n] for n<47, else 0   (pad N to 48)
__global__ void conv_w2(const float* __restrict__ W2, half_t* __restrict__ W2hT) {
  int i = blockIdx.x * blockDim.x + threadIdx.x;   // 48*128
  int k = i & 127, n = i >> 7;
  W2hT[n * HDIM + k] = (n < CDIM) ? (half_t)W2[k * CDIM + n] : (half_t)0.0f;
}

// degree histograms: deg_src[esrc[i]] += 1 ; deg_dst[edst[i]] += 1
__global__ void deg_hist(const int* __restrict__ esrc, const int* __restrict__ edst,
                         float* __restrict__ dsrc, float* __restrict__ ddst, int E) {
  int i = blockIdx.x * blockDim.x + threadIdx.x;
  if (i < E) {
    atomicAdd(dsrc + esrc[i], 1.0f);
    atomicAdd(ddst + edst[i], 1.0f);
  }
}

// aggX[dst] += embed[inputs[src]] * rsqrt(max(deg_out0[src],1)) ; 32 threads/edge, 4 floats each
__global__ void edge_agg1(const int* __restrict__ e_src, const int* __restrict__ e_dst,
                          const long long* __restrict__ inputs,
                          const float* __restrict__ embed,
                          const float* __restrict__ deg_out0,
                          float* __restrict__ aggX, int E) {
  int tid = blockIdx.x * blockDim.x + threadIdx.x;
  int e = tid >> 5;
  if (e >= E) return;
  int c = (tid & 31) << 2;
  int s = e_src[e], d = e_dst[e];
  long long node = inputs[s];
  float sc = rsqrtf(fmaxf(deg_out0[s], 1.0f));
  float4 x = *reinterpret_cast<const float4*>(embed + node * (long long)HDIM + c);
  float* o = aggX + (long long)d * HDIM + c;
  atomicAdd(o + 0, x.x * sc);
  atomicAdd(o + 1, x.y * sc);
  atomicAdd(o + 2, x.z * sc);
  atomicAdd(o + 3, x.w * sc);
}

// ---------------------------------------------------------------- WMMA GEMMs
// Fragment layouts per CDNA5 ISA 7.12.2 (wave32):
//  A 16x32 f16 : lane r=l&15 -> row M=r, group g=l>>4;
//                vgpr v<4 : K = g*8+2v,+1 ; v>=4 : K = 16+g*8+2(v-4),+1
//  B 32x16 f16 : lane r -> col N=r ; vgpr v : K same as A (g*8+2v,+1 ; 16+g*8+2(v-4),+1)
//  C/D 16x16 f32: lane r -> N=r ; vgpr v -> M = v + 8*g

// h1h = f16( rout1[m] * relu( (rin0[m]*aggX[m]) @ W1 + b1 ) ),  M=N1, N=K=128
__global__ __launch_bounds__(256)
void gemm1_wmma(const float* __restrict__ aggX, const half_t* __restrict__ W1hT,
                const float* __restrict__ b1,
                const float* __restrict__ deg_in0, const float* __restrict__ deg_out1,
                half_t* __restrict__ h1h) {
  const int lane = threadIdx.x & 31;
  const int tile = blockIdx.x * 8 + (threadIdx.x >> 5);
  const int mt = tile >> 3;            // 8 N-tiles of 16
  const int nt = tile & 7;
  const int r = lane & 15, g = lane >> 4;

  const int rowA = mt * 16 + r;
  const int colB = nt * 16 + r;
  const float sA = rsqrtf(fmaxf(deg_in0[rowA], 1.0f));

  v8f c = {};
  for (int kk = 0; kk < HDIM; kk += 32) {
    v16h a, b;
#pragma unroll
    for (int v = 0; v < 8; ++v) {
      int ka = kk + ((v < 4) ? (g * 8 + 2 * v) : (16 + g * 8 + 2 * (v - 4)));
      float2 x = *reinterpret_cast<const float2*>(aggX + rowA * HDIM + ka);
      a[2 * v]     = (half_t)(x.x * sA);
      a[2 * v + 1] = (half_t)(x.y * sA);
      int kb = kk + ((v < 4) ? (g * 8 + 2 * v) : (16 + g * 8 + 2 * (v - 4)));
      h2 bb = *reinterpret_cast<const h2*>(W1hT + colB * HDIM + kb);
      b[2 * v]     = bb.x;
      b[2 * v + 1] = bb.y;
    }
    c = __builtin_amdgcn_wmma_f32_16x16x32_f16(false, a, false, b, (short)0, c,
                                               false, false);
  }
  const float bn = b1[colB];
#pragma unroll
  for (int v = 0; v < 8; ++v) {
    int m = mt * 16 + v + 8 * g;
    float val = fmaxf(c[v] + bn, 0.0f);
    val *= rsqrtf(fmaxf(deg_out1[m], 1.0f));
    h1h[m * HDIM + colB] = (half_t)val;
  }
}

// tmp2 = h1h @ W2hT^T  (f32 out),  M=N1, K=128, N=48 (cols 47 are zero via W pad)
__global__ __launch_bounds__(256)
void gemm2_wmma(const half_t* __restrict__ h1h, const half_t* __restrict__ W2hT,
                float* __restrict__ tmp2) {
  const int lane = threadIdx.x & 31;
  const int tile = blockIdx.x * 8 + (threadIdx.x >> 5);
  const int mt = tile / 3;             // 3 N-tiles of 16
  const int nt = tile % 3;
  const int r = lane & 15, g = lane >> 4;

  const int rowA = mt * 16 + r;
  const int colB = nt * 16 + r;

  v8f c = {};
  for (int kk = 0; kk < HDIM; kk += 32) {
    v16h a, b;
#pragma unroll
    for (int v = 0; v < 8; ++v) {
      int ka = kk + ((v < 4) ? (g * 8 + 2 * v) : (16 + g * 8 + 2 * (v - 4)));
      h2 aa = *reinterpret_cast<const h2*>(h1h + rowA * HDIM + ka);
      a[2 * v]     = aa.x;
      a[2 * v + 1] = aa.y;
      int kb = kk + ((v < 4) ? (g * 8 + 2 * v) : (16 + g * 8 + 2 * (v - 4)));
      h2 bb = *reinterpret_cast<const h2*>(W2hT + colB * HDIM + kb);
      b[2 * v]     = bb.x;
      b[2 * v + 1] = bb.y;
    }
    c = __builtin_amdgcn_wmma_f32_16x16x32_f16(false, a, false, b, (short)0, c,
                                               false, false);
  }
#pragma unroll
  for (int v = 0; v < 8; ++v) {
    int m = mt * 16 + v + 8 * g;
    tmp2[m * CPAD + colB] = c[v];
  }
}

// ---------------------------------------------------------------- layer 2 agg + finalize

// outacc[dst*48+c] += tmp2[src*48+c] ; 12 threads/edge, 4 floats each
__global__ void edge_agg2(const int* __restrict__ e_src, const int* __restrict__ e_dst,
                          const float* __restrict__ tmp2, float* __restrict__ outacc,
                          int E) {
  int tid = blockIdx.x * blockDim.x + threadIdx.x;
  int e = tid / 12;
  if (e >= E) return;
  int c = (tid % 12) << 2;
  int s = e_src[e], d = e_dst[e];
  float4 x = *reinterpret_cast<const float4*>(tmp2 + s * CPAD + c);
  float* o = outacc + d * CPAD + c;
  atomicAdd(o + 0, x.x);
  atomicAdd(o + 1, x.y);
  atomicAdd(o + 2, x.z);
  atomicAdd(o + 3, x.w);
}

__global__ void finalize(const float* __restrict__ outacc, const float* __restrict__ deg_in1,
                         const float* __restrict__ b2, float* __restrict__ out) {
  int i = blockIdx.x * blockDim.x + threadIdx.x;
  if (i < N2 * CDIM) {
    int n = i / CDIM, c = i % CDIM;
    out[i] = outacc[n * CPAD + c] * rsqrtf(fmaxf(deg_in1[n], 1.0f)) + b2[c];
  }
}

// ---------------------------------------------------------------- launch

extern "C" void kernel_launch(void* const* d_in, const int* in_sizes, int n_in,
                              void* d_out, int out_size, void* d_ws, size_t ws_size,
                              hipStream_t stream) {
  const float*     embed  = (const float*)d_in[0];
  const float*     W1     = (const float*)d_in[1];
  const float*     b1     = (const float*)d_in[2];
  const float*     W2     = (const float*)d_in[3];
  const float*     b2     = (const float*)d_in[4];
  const long long* inputs = (const long long*)d_in[5];
  const int*       e0_src = (const int*)d_in[6];
  const int*       e0_dst = (const int*)d_in[7];
  const int*       e1_src = (const int*)d_in[8];
  const int*       e1_dst = (const int*)d_in[9];
  const int E0 = in_sizes[6];
  const int E1 = in_sizes[8];
  float* out = (float*)d_out;

  // workspace layout (all 256B aligned); zeroed region is contiguous at the front
  char* ws = (char*)d_ws;
  float*  aggX     = (float*)(ws);                         // N1*128*4 = 33554432
  float*  outacc   = (float*)(ws + 33554432);              // N2*48*4  =  3145728
  float*  deg_out0 = (float*)(ws + 36700160);              // N0*4     =  1048576
  float*  deg_in0  = (float*)(ws + 37748736);              // N1*4     =   262144
  float*  deg_out1 = (float*)(ws + 38010880);              // N1*4     =   262144
  float*  deg_in1  = (float*)(ws + 38273024);              // N2*4     =    65536
  const int zeroFloats = 38338560 / 4;                     // everything above
  half_t* h1h      = (half_t*)(ws + 38338560);             // N1*128*2 = 16777216
  float*  tmp2     = (float*)(ws + 55115776);              // N1*48*4  = 12582912
  half_t* W1hT     = (half_t*)(ws + 67698688);             // 128*128*2
  half_t* W2hT     = (half_t*)(ws + 67731456);             // 48*128*2

  const int T = 256;

  // 1. clear accumulators + degree histograms
  zero_f32<<<(zeroFloats + T - 1) / T, T, 0, stream>>>((float*)ws, zeroFloats);
  // 2. weight conversion (f32 -> f16, transposed / padded)
  conv_w1<<<(HDIM * HDIM + T - 1) / T, T, 0, stream>>>(W1, W1hT);
  conv_w2<<<(CPAD * HDIM + T - 1) / T, T, 0, stream>>>(W2, W2hT);
  // 3. degrees
  deg_hist<<<(E0 + T - 1) / T, T, 0, stream>>>(e0_src, e0_dst, deg_out0, deg_in0, E0);
  deg_hist<<<(E1 + T - 1) / T, T, 0, stream>>>(e1_src, e1_dst, deg_out1, deg_in1, E1);
  // 4. layer-1 aggregation of scaled, gathered embeddings (aggregate-then-project)
  edge_agg1<<<((long long)E0 * 32 + T - 1) / T, T, 0, stream>>>(
      e0_src, e0_dst, inputs, embed, deg_out0, aggX, E0);
  // 5. WMMA GEMM1 fused with deg_in scale, bias, relu, layer-2 deg_out scale, f16 cast
  gemm1_wmma<<<(N1 / 16) * 8 / 8, T, 0, stream>>>(aggX, W1hT, b1, deg_in0, deg_out1, h1h);
  // 6. WMMA GEMM2 (N padded to 48)
  gemm2_wmma<<<(N1 / 16) * 3 / 8, T, 0, stream>>>(h1h, W2hT, tmp2);
  // 7. layer-2 aggregation
  edge_agg2<<<((long long)E1 * 12 + T - 1) / T, T, 0, stream>>>(e1_src, e1_dst, tmp2,
                                                                outacc, E1);
  // 8. finalize: deg_in scale + bias
  finalize<<<(N2 * CDIM + T - 1) / T, T, 0, stream>>>(outacc, deg_in1, b2, out);
}
